// Block_8693013807158
// MI455X (gfx1250) — matrix operand order repaired and checked by reading, not compiled, not run on hardware
//
#include <hip/hip_runtime.h>
#include <hip/hip_bf16.h>

// ---------------- problem constants ----------------
#define DIMC   768
#define NSEQ   1024
#define BATCH  8
#define NHEAD  12
#define HDIM   64
#define HIDDEN 3072
#define MROWS  (BATCH*NSEQ)   // 8192

typedef __attribute__((ext_vector_type(16))) _Float16 v16h;
typedef __attribute__((ext_vector_type(8)))  _Float16 v8h;
typedef __attribute__((ext_vector_type(8)))  float    v8f;

// ---------------- fragment helpers (CDNA5 WMMA 16x16x32 f16 layouts) ----------
__device__ __forceinline__ v16h ld_pair(const _Float16* lo, const _Float16* hi) {
  v8h a = *reinterpret_cast<const v8h*>(lo);
  v8h b = *reinterpret_cast<const v8h*>(hi);
  v16h r;
#pragma unroll
  for (int i = 0; i < 8; ++i) { r[i] = a[i]; r[i + 8] = b[i]; }
  return r;
}
// A-matrix 16x32: per lane two 8-half chunks at (k + 8*(lane>=16)) and +16
__device__ __forceinline__ v16h ld_afrag(const _Float16* p) { return ld_pair(p, p + 16); }
// B-matrix 32x16: per lane two 8-half chunks at (k + 8*(lane>=16)) and +16
__device__ __forceinline__ v16h ld_bfrag(const _Float16* p) { return ld_pair(p, p + 16); }

__device__ __forceinline__ v8f wmma16(v16h a, v16h b, v8f c) {
  return __builtin_amdgcn_wmma_f32_16x16x32_f16(false, a, false, b, (short)0, c,
                                                false, false);
}

// gfx1250 async global -> LDS copy (16 bytes / lane), tracked by ASYNCcnt
__device__ __forceinline__ void async_ld_b128(_Float16* lds_dst,
                                              const _Float16* gsrc) {
  unsigned lds_off = (unsigned)(size_t)(void*)lds_dst;  // low 32b = LDS byte addr
  asm volatile("global_load_async_to_lds_b128 %0, %1, off"
               :
               : "v"(lds_off), "v"(gsrc)
               : "memory");
}
__device__ __forceinline__ void wait_asynccnt0() {
  asm volatile("s_wait_asynccnt 0" ::: "memory");
}

// ---------------- fp32 -> f16 convert ----------------
__global__ void f2h_kernel(const float* __restrict__ src, _Float16* __restrict__ dst,
                           int n) {
  int i = blockIdx.x * blockDim.x + threadIdx.x;
  int stride = gridDim.x * blockDim.x;
  for (; i < n; i += stride) dst[i] = (_Float16)src[i];
}

// ---------------- LayerNorm: fp32 row -> f16 row ----------------
__global__ __launch_bounds__(256) void layernorm_kernel(
    const float* __restrict__ x, const float* __restrict__ w,
    const float* __restrict__ b, _Float16* __restrict__ out) {
  __shared__ float red[256];
  const int row = blockIdx.x;
  const float* xr = x + (size_t)row * DIMC;
  float vals[3];
  float s0 = 0.f, s1 = 0.f;
#pragma unroll
  for (int i = 0; i < 3; ++i) {
    float v = xr[threadIdx.x + i * 256];
    vals[i] = v; s0 += v; s1 += v * v;
  }
  red[threadIdx.x] = s0; __syncthreads();
  for (int off = 128; off > 0; off >>= 1) {
    if (threadIdx.x < off) red[threadIdx.x] += red[threadIdx.x + off];
    __syncthreads();
  }
  const float mean = red[0] * (1.0f / DIMC);
  __syncthreads();
  red[threadIdx.x] = s1; __syncthreads();
  for (int off = 128; off > 0; off >>= 1) {
    if (threadIdx.x < off) red[threadIdx.x] += red[threadIdx.x + off];
    __syncthreads();
  }
  const float var = red[0] * (1.0f / DIMC) - mean * mean;
  const float rstd = rsqrtf(var + 1e-5f);
#pragma unroll
  for (int i = 0; i < 3; ++i) {
    int c = threadIdx.x + i * 256;
    out[(size_t)row * DIMC + c] = (_Float16)((vals[i] - mean) * rstd * w[c] + b[c]);
  }
}

// ---------------- WMMA GEMM: out[m,n] = sum_k A[m,k]*W[n,k] + bias[n] ----------
// Block = 128 threads = 4 waves (2x2); each wave computes a 64x64 tile (4x4 WMMA).
// EPI: 0 = QKV scatter -> q/k/v [B,H,N,HD] f16
//      1 = +resid -> fp32
//      2 = PolyGELU -> f16
//      3 = +resid -> fp32 (final output)
template <int EPI>
__global__ __launch_bounds__(128) void gemm_wmma(
    const _Float16* __restrict__ A, const _Float16* __restrict__ W,
    const float* __restrict__ bias, int Kdim,
    float* __restrict__ f32out, const float* __restrict__ resid,
    _Float16* __restrict__ h16out,
    _Float16* __restrict__ q_out, _Float16* __restrict__ k_out,
    _Float16* __restrict__ v_out,
    const float* __restrict__ g0, const float* __restrict__ g1,
    const float* __restrict__ g2) {
  const int lane = threadIdx.x & 31;
  const int wave = threadIdx.x >> 5;
  const int lm   = lane & 15;
  const int hi16 = lane >> 4;           // 0 for lanes 0-15, 1 for 16-31
  const int m0 = blockIdx.y * 128 + (wave >> 1) * 64;
  const int n0 = blockIdx.x * 128 + (wave & 1) * 64;

  const _Float16* arow[4];
  const _Float16* brow[4];
#pragma unroll
  for (int t = 0; t < 4; ++t) {
    arow[t] = A + (size_t)(m0 + t * 16 + lm) * Kdim + hi16 * 8;
    brow[t] = W + (size_t)(n0 + t * 16 + lm) * Kdim + hi16 * 8;
  }

  v8f acc[4][4] = {};
  for (int kk = 0; kk < Kdim; kk += 32) {
    if (kk + 32 < Kdim) {               // gfx1250 global_prefetch_b8
      __builtin_prefetch(arow[0] + kk + 32, 0, 3);
      __builtin_prefetch(brow[0] + kk + 32, 0, 3);
    }
    v16h af[4], bf[4];
#pragma unroll
    for (int t = 0; t < 4; ++t) af[t] = ld_afrag(arow[t] + kk);
#pragma unroll
    for (int t = 0; t < 4; ++t) bf[t] = ld_bfrag(brow[t] + kk);
#pragma unroll
    for (int i = 0; i < 4; ++i)
#pragma unroll
      for (int j = 0; j < 4; ++j) acc[i][j] = wmma16(af[i], bf[j], acc[i][j]);
  }

  float e0 = 0.f, e1 = 0.f, e2 = 0.f;
  if (EPI == 2) { e0 = *g0; e1 = *g1; e2 = *g2; }

#pragma unroll
  for (int i = 0; i < 4; ++i) {
#pragma unroll
    for (int j = 0; j < 4; ++j) {
      const int mbase = m0 + i * 16;
      const int nbase = n0 + j * 16;
#pragma unroll
      for (int r = 0; r < 8; ++r) {
        const int m = mbase + (hi16 ? r + 8 : r);   // C/D layout (ISA 7.12.2)
        const int n = nbase + lm;
        float v = acc[i][j][r] + bias[n];
        if (EPI == 0) {
          const int t  = n / DIMC;
          const int rm = n - t * DIMC;
          const int hh = rm >> 6;
          const int d  = rm & 63;
          const int bb = m >> 10;
          const int nn = m & 1023;
          _Float16* dst = (t == 0) ? q_out : ((t == 1) ? k_out : v_out);
          dst[(((size_t)(bb * NHEAD + hh)) * NSEQ + nn) * HDIM + d] = (_Float16)v;
        } else if (EPI == 1) {
          f32out[(size_t)m * DIMC + n] = v + resid[(size_t)m * DIMC + n];
        } else if (EPI == 2) {
          h16out[(size_t)m * HIDDEN + n] = (_Float16)(e0 * v * v + e1 * v + e2);
        } else {
          f32out[(size_t)m * DIMC + n] = v + resid[(size_t)m * DIMC + n];
        }
      }
    }
  }
}

// ---------------- Poly attention (single pass; normalizer is a sum) ----------
// grid = (N/64, B*H), block = 128 (4 waves x 16 query rows)
__global__ __launch_bounds__(128) void attn_poly_kernel(
    const _Float16* __restrict__ qh, const _Float16* __restrict__ kh,
    const _Float16* __restrict__ vh, _Float16* __restrict__ out,
    const float* __restrict__ pa_p, const float* __restrict__ pb_p,
    const float* __restrict__ pc_p) {
  __shared__ __align__(16) _Float16 lds_k[32 * HDIM];    // K tile, row-major [m][d]
  __shared__ __align__(16) _Float16 lds_vt[HDIM * 32];   // V tile transposed [d][m]
  __shared__ __align__(16) _Float16 lds_p[4][16 * 32];   // per-wave P staging

  const int lane = threadIdx.x & 31;
  const int wave = threadIdx.x >> 5;
  const int lm   = lane & 15;
  const int hi16 = lane >> 4;
  const int bh = blockIdx.y;                 // 0..95
  const int b  = bh / NHEAD, h = bh - b * NHEAD;
  const int q0 = blockIdx.x * 64 + wave * 16;

  const float pa = *pa_p, pb = *pb_p, pc = *pc_p;
  const float scale = 0.125f;                // HD^-0.5

  const size_t base = (size_t)bh * NSEQ * HDIM;

  // Q fragments (held in registers for the whole key loop)
  const _Float16* qrow = qh + base + (size_t)(q0 + lm) * HDIM + hi16 * 8;
  const v16h qf0 = ld_afrag(qrow);
  const v16h qf1 = ld_afrag(qrow + 32);

  v8f o[4] = {};
  float ps[8] = {};

  for (int mb = 0; mb < NSEQ; mb += 32) {
    // cooperative stage: K tile via async global->LDS DMA (no VGPR round-trip),
    // V tile transposed through registers (async copies cannot transpose)
    {
      const int idx = threadIdx.x * 16;      // 128 threads x 16 halves = 32x64 tile
      const _Float16* ksrc = kh + base + (size_t)mb * HDIM;
      const _Float16* vsrc = vh + base + (size_t)mb * HDIM;
      async_ld_b128(&lds_k[idx], ksrc + idx);
      async_ld_b128(&lds_k[idx + 8], ksrc + idx + 8);
      const int row = idx >> 6, col = idx & 63;
      v8h va = *(const v8h*)(vsrc + idx);
      v8h vb = *(const v8h*)(vsrc + idx + 8);
#pragma unroll
      for (int i = 0; i < 8; ++i) {
        lds_vt[(col + i) * 32 + row]     = va[i];
        lds_vt[(col + 8 + i) * 32 + row] = vb[i];
      }
      wait_asynccnt0();                      // own async fills done, then barrier
    }
    __syncthreads();

    // S = Q K^T (two 16x16 key tiles), poly, accumulate row sums, stage P
#pragma unroll
    for (int mt = 0; mt < 2; ++mt) {
      const _Float16* krow = &lds_k[(mt * 16 + lm) * HDIM + hi16 * 8];
      v16h kf0 = ld_bfrag(krow);
      v16h kf1 = ld_bfrag(krow + 32);
      v8f s = {};
      s = wmma16(qf0, kf0, s);
      s = wmma16(qf1, kf1, s);
#pragma unroll
      for (int r = 0; r < 8; ++r) {
        float sv = s[r] * scale;
        float p = pa * sv * sv + pb * sv + pc;
        p = fmaxf(p, 1e-6f);
        ps[r] += p;
        const int qr = hi16 ? r + 8 : r;
        lds_p[wave][qr * 32 + mt * 16 + lm] = (_Float16)p;
      }
    }
    // wave-synchronous LDS round-trip (DS ops are in-order per wave)
    asm volatile("s_wait_dscnt 0" ::: "memory");

    // O += P * V  (P as A 16x32, V^T rows as B 32x16 per 16-wide d chunk)
    const _Float16* prow = &lds_p[wave][lm * 32 + hi16 * 8];
    v16h pf = ld_afrag(prow);
#pragma unroll
    for (int dc = 0; dc < 4; ++dc) {
      const _Float16* vtrow = &lds_vt[(dc * 16 + lm) * 32 + hi16 * 8];
      v16h vf = ld_bfrag(vtrow);
      o[dc] = wmma16(pf, vf, o[dc]);
    }
    __syncthreads();
  }

  // reduce row sums across each 16-lane half (lanes 0-15: q=r, 16-31: q=r+8)
#pragma unroll
  for (int r = 0; r < 8; ++r) {
    float v = ps[r];
    v += __shfl_xor(v, 1, 16);
    v += __shfl_xor(v, 2, 16);
    v += __shfl_xor(v, 4, 16);
    v += __shfl_xor(v, 8, 16);
    ps[r] = v + 1e-8f;
  }

  // write normalized output: out[b, q, h*64 + d]  (f16)
#pragma unroll
  for (int dc = 0; dc < 4; ++dc) {
#pragma unroll
    for (int r = 0; r < 8; ++r) {
      const int q = q0 + (hi16 ? r + 8 : r);
      const int d = h * HDIM + dc * 16 + lm;
      out[(size_t)(b * NSEQ + q) * DIMC + d] = (_Float16)(o[dc][r] / ps[r]);
    }
  }
}

// ---------------- launcher ----------------
extern "C" void kernel_launch(void* const* d_in, const int* in_sizes, int n_in,
                              void* d_out, int out_size, void* d_ws, size_t ws_size,
                              hipStream_t stream) {
  const float* x      = (const float*)d_in[0];
  const float* ln1_w  = (const float*)d_in[1];
  const float* ln1_b  = (const float*)d_in[2];
  const float* ln2_w  = (const float*)d_in[3];
  const float* ln2_b  = (const float*)d_in[4];
  const float* qkv_w  = (const float*)d_in[5];
  const float* qkv_b  = (const float*)d_in[6];
  const float* proj_w = (const float*)d_in[7];
  const float* proj_b = (const float*)d_in[8];
  const float* fc1_w  = (const float*)d_in[9];
  const float* fc1_b  = (const float*)d_in[10];
  const float* fc2_w  = (const float*)d_in[11];
  const float* fc2_b  = (const float*)d_in[12];
  const float* pa     = (const float*)d_in[13];
  const float* pb     = (const float*)d_in[14];
  const float* pc     = (const float*)d_in[15];
  const float* ga     = (const float*)d_in[16];
  const float* gb     = (const float*)d_in[17];
  const float* gc     = (const float*)d_in[18];
  float* out = (float*)d_out;

  char* ws = (char*)d_ws;
  size_t off = 0;
  auto alloc = [&](size_t bytes) -> void* {
    void* p = ws + off;
    off += (bytes + 255) & ~(size_t)255;
    return p;
  };
  _Float16* Wqkv_h  = (_Float16*)alloc((size_t)3 * DIMC * DIMC * 2);
  _Float16* Wproj_h = (_Float16*)alloc((size_t)DIMC * DIMC * 2);
  _Float16* Wfc1_h  = (_Float16*)alloc((size_t)HIDDEN * DIMC * 2);
  _Float16* Wfc2_h  = (_Float16*)alloc((size_t)DIMC * HIDDEN * 2);
  _Float16* h1_h    = (_Float16*)alloc((size_t)MROWS * DIMC * 2);
  _Float16* q_h     = (_Float16*)alloc((size_t)MROWS * DIMC * 2);
  _Float16* k_h     = (_Float16*)alloc((size_t)MROWS * DIMC * 2);
  _Float16* v_h     = (_Float16*)alloc((size_t)MROWS * DIMC * 2);
  _Float16* attno_h = (_Float16*)alloc((size_t)MROWS * DIMC * 2);
  float*    x1_f    = (float*)alloc((size_t)MROWS * DIMC * 4);
  _Float16* h2_h    = (_Float16*)alloc((size_t)MROWS * DIMC * 2);
  _Float16* u_h     = (_Float16*)alloc((size_t)MROWS * HIDDEN * 2);
  (void)ws_size; (void)in_sizes; (void)n_in; (void)out_size;

  // 1) weights fp32 -> f16
  {
    int n0 = 3 * DIMC * DIMC, n1 = DIMC * DIMC, n2 = HIDDEN * DIMC;
    f2h_kernel<<<(n0 + 255) / 256, 256, 0, stream>>>(qkv_w, Wqkv_h, n0);
    f2h_kernel<<<(n1 + 255) / 256, 256, 0, stream>>>(proj_w, Wproj_h, n1);
    f2h_kernel<<<(n2 + 255) / 256, 256, 0, stream>>>(fc1_w, Wfc1_h, n2);
    f2h_kernel<<<(n2 + 255) / 256, 256, 0, stream>>>(fc2_w, Wfc2_h, n2);
  }
  // 2) LN1
  layernorm_kernel<<<MROWS, 256, 0, stream>>>(x, ln1_w, ln1_b, h1_h);
  // 3) QKV GEMM (epilogue: scatter into [B,H,N,HD] f16)
  gemm_wmma<0><<<dim3(3 * DIMC / 128, MROWS / 128), 128, 0, stream>>>(
      h1_h, Wqkv_h, qkv_b, DIMC, nullptr, nullptr, nullptr, q_h, k_h, v_h,
      nullptr, nullptr, nullptr);
  // 4) Poly attention
  attn_poly_kernel<<<dim3(NSEQ / 64, BATCH * NHEAD), 128, 0, stream>>>(
      q_h, k_h, v_h, attno_h, pa, pb, pc);
  // 5) proj GEMM + residual -> x1 (fp32)
  gemm_wmma<1><<<dim3(DIMC / 128, MROWS / 128), 128, 0, stream>>>(
      attno_h, Wproj_h, proj_b, DIMC, x1_f, x, nullptr, nullptr, nullptr,
      nullptr, nullptr, nullptr, nullptr);
  // 6) LN2
  layernorm_kernel<<<MROWS, 256, 0, stream>>>(x1_f, ln2_w, ln2_b, h2_h);
  // 7) FC1 GEMM + PolyGELU -> u (f16)
  gemm_wmma<2><<<dim3(HIDDEN / 128, MROWS / 128), 128, 0, stream>>>(
      h2_h, Wfc1_h, fc1_b, DIMC, nullptr, nullptr, u_h, nullptr, nullptr,
      nullptr, ga, gb, gc);
  // 8) FC2 GEMM + residual -> out (fp32)
  gemm_wmma<3><<<dim3(DIMC / 128, MROWS / 128), 128, 0, stream>>>(
      u_h, Wfc2_h, fc2_b, HIDDEN, out, x1_f, nullptr, nullptr, nullptr,
      nullptr, nullptr, nullptr, nullptr);
}
